// Sequence_82051055222834
// MI455X (gfx1250) — matrix operand order repaired and checked by reading, not compiled, not run on hardware
//
#include <hip/hip_runtime.h>
#include <hip/hip_bf16.h>
#include <cstdint>

// ---------------------------------------------------------------------------
// 2-layer GRU, persistent-workgroup kernel for gfx1250 (MI455X).
// bf16 WMMA (v_wmma_f32_16x16x32_bf16), f32 accumulate, LDS-resident state.
// All GEMM work (including the K=8 input projection, zero-padded to K=32
// with a constant-1 bias column) runs on the WMMA path.
// ---------------------------------------------------------------------------

typedef __bf16 bf16_t;
typedef __attribute__((ext_vector_type(16))) __bf16 v16bf;
typedef __attribute__((ext_vector_type(8)))  __bf16 v8bf;
typedef __attribute__((ext_vector_type(8)))  float  v8f;

#define B_TOT 2048
#define T_TOT 128
#define HDIM  512
#define GDIM  1536                 // 3*H
#define NOUT  5
#define NCMD  3
#define IND   8                    // NOUT + NCMD
#define XSTR  32                   // padded x row width (K=32 WMMA tile)
#define BTILE 32                   // batch rows per workgroup
#define MT    (BTILE/16)           // 2 m-tiles
#define KT    (HDIM/32)            // 16 k-steps of 32
#define CCH   (HDIM/16)            // 32 hidden-column chunks
#define NTT   (GDIM/16)            // 96 packed n-tiles per weight
#define NWAVE 8
#define NTHREADS (NWAVE*32)
#define NUNITS (MT*CCH)            // 64 work units per layer

static constexpr size_t PW_ELEMS    = (size_t)KT*NTT*32*16; // per big weight
static constexpr size_t PWOUT_ELEMS = (size_t)KT*32*16;
static constexpr size_t PWIH1_ELEMS = (size_t)NTT*32*16;    // single k-tile

__device__ __forceinline__ v8f wmma_bf16(v16bf a, v16bf b, v8f c) {
  return __builtin_amdgcn_wmma_f32_16x16x32_bf16(
      /*neg_a=*/false, a, /*neg_b=*/false, b,
      /*c_mod=*/(short)0, c, /*reuse_a=*/false, /*reuse_b=*/false);
}

__device__ __forceinline__ float sigmoidf_(float x) {
  return 1.0f / (1.0f + __expf(-x));
}
__device__ __forceinline__ float tanhf_(float x) {
  float e = __expf(-2.0f * x);
  return (1.0f - e) / (1.0f + e);
}

// Load one 16x32 bf16 A-fragment from an LDS tile with row stride `stride`.
// Lane half h holds K = kbase + 16*(lane>=16) + h.
__device__ __forceinline__ v16bf load_afrag(const bf16_t* base, int row,
                                            int stride, int kbase, int sel) {
  union { v16bf v; v8bf h8[2]; } af;
  const bf16_t* p = base + row * stride + kbase + 16 * sel;
  af.h8[0] = *(const v8bf*)(p);
  af.h8[1] = *(const v8bf*)(p + 8);
  return af.v;
}

// --- weight packing: W[G][K] f32 row-major -> bf16 WMMA B-fragment order ----
// dst layout: [kt][nt][lane][16 halves]; lane half h holds
//   K = kt*32 + 16*(lane>=16) + h,  N(gate col) = nt*16 + (lane&15)
__global__ void pack_weight_kernel(const float* __restrict__ W,
                                   bf16_t* __restrict__ dst, int Kdim) {
  int idx = blockIdx.x * blockDim.x + threadIdx.x;
  const int tot = KT * NTT * 32;
  if (idx >= tot) return;
  int lane = idx & 31;
  int nt   = (idx >> 5) % NTT;
  int kt   = (idx >> 5) / NTT;
  int g  = nt * 16 + (lane & 15);
  int kb = kt * 32 + ((lane >> 4) << 4);
  const float* src = W + (size_t)g * Kdim + kb;
  bf16_t* o = dst + (size_t)idx * 16;
#pragma unroll
  for (int h = 0; h < 16; ++h) o[h] = (bf16_t)src[h];
}

// W_ih1 (1536 x 8) + b_ih1 -> one zero-padded K=32 k-tile.
// Row k<8: W_ih1[g][k]; row k==8: b_ih1[g] (multiplies the constant-1 x col).
__global__ void pack_wih1_kernel(const float* __restrict__ W,
                                 const float* __restrict__ bias,
                                 bf16_t* __restrict__ dst) {
  int idx = blockIdx.x * blockDim.x + threadIdx.x;
  const int tot = NTT * 32;
  if (idx >= tot) return;
  int lane = idx & 31;
  int nt   = idx >> 5;
  int g = nt * 16 + (lane & 15);
  bf16_t* o = dst + (size_t)idx * 16;
#pragma unroll
  for (int h = 0; h < 16; ++h) {
    int k = ((lane >> 4) << 4) + h;
    float v = (k < IND) ? W[(size_t)g * IND + k]
            : (k == IND) ? bias[g] : 0.0f;
    o[h] = (bf16_t)v;
  }
}

// W_out (5 x 512) -> one zero-padded 16-wide packed n-tile
__global__ void pack_wout_kernel(const float* __restrict__ W,
                                 bf16_t* __restrict__ dst) {
  int idx = blockIdx.x * blockDim.x + threadIdx.x;
  const int tot = KT * 32;
  if (idx >= tot) return;
  int lane = idx & 31;
  int kt   = idx >> 5;
  int g  = lane & 15;
  int kb = kt * 32 + ((lane >> 4) << 4);
  bf16_t* o = dst + (size_t)idx * 16;
#pragma unroll
  for (int h = 0; h < 16; ++h)
    o[h] = (g < NOUT) ? (bf16_t)W[(size_t)g * HDIM + kb + h] : (bf16_t)0.0f;
}

// --- main persistent GRU kernel --------------------------------------------
__global__ __launch_bounds__(NTHREADS)
void gru_persistent(const float* __restrict__ cmds,
                    const float* __restrict__ y0,
                    const float* __restrict__ h0_1,
                    const float* __restrict__ h0_2,
                    const float* __restrict__ b_hh1,
                    const float* __restrict__ b_ih2,
                    const float* __restrict__ b_hh2,
                    const float* __restrict__ b_out,
                    const bf16_t* __restrict__ pWih1,
                    const bf16_t* __restrict__ pWhh1,
                    const bf16_t* __restrict__ pWih2,
                    const bf16_t* __restrict__ pWhh2,
                    const bf16_t* __restrict__ pWout,
                    float* __restrict__ out) {
  extern __shared__ char smem[];
  bf16_t* h1buf = (bf16_t*)smem;                       // 2 * BTILE*HDIM bf16
  bf16_t* h2buf = h1buf + 2 * BTILE * HDIM;            // 2 * BTILE*HDIM bf16
  bf16_t* xbuf  = h2buf + 2 * BTILE * HDIM;            // BTILE * XSTR bf16

  const int tid   = threadIdx.x;
  const int lane  = tid & 31;
  const int wave  = tid >> 5;
  const int bbase = blockIdx.x * BTILE;
  const int n     = lane & 15;   // N column within 16-tile (C/D layout)
  const int sel   = lane >> 4;   // lane half
  const int mA    = lane & 15;   // A-fragment row

  // ---- init state tiles (buffer 0) and padded x(t=0) ----
  for (int i = tid; i < BTILE * HDIM; i += NTHREADS) {
    int r = i / HDIM, c = i % HDIM;
    h1buf[i] = (bf16_t)h0_1[(size_t)(bbase + r) * HDIM + c];
    h2buf[i] = (bf16_t)h0_2[(size_t)(bbase + r) * HDIM + c];
  }
  for (int i = tid; i < BTILE * XSTR; i += NTHREADS) {
    int r = i >> 5, c = i & 31;
    float v = 0.0f;
    if (c < NOUT)          v = y0[(size_t)(bbase + r) * NOUT + c];
    else if (c < IND)      v = cmds[(size_t)(bbase + r) * NCMD + (c - NOUT)];
    else if (c == IND)     v = 1.0f;   // constant-1 bias column
    xbuf[i] = (bf16_t)v;
  }
  __syncthreads();

  const v16bf* bp_ih1 = (const v16bf*)pWih1;
  const v16bf* bp_hh1 = (const v16bf*)pWhh1;
  const v16bf* bp_ih2 = (const v16bf*)pWih2;
  const v16bf* bp_hh2 = (const v16bf*)pWhh2;
  const v16bf* bp_out = (const v16bf*)pWout;

  for (int t = 0; t < T_TOT; ++t) {
    const bf16_t* h1c = h1buf + (t & 1) * BTILE * HDIM;
    bf16_t*       h1n = h1buf + ((t + 1) & 1) * BTILE * HDIM;
    const bf16_t* h2c = h2buf + (t & 1) * BTILE * HDIM;
    bf16_t*       h2n = h2buf + ((t + 1) & 1) * BTILE * HDIM;

    // ================= layer 1 =================
    for (int u = wave; u < NUNITS; u += NWAVE) {
      const int mt   = u & (MT - 1);
      const int c    = u / MT;          // hidden column chunk
      const int m0   = mt * 16;
      const int gcol = c * 16 + n;      // 0..511

      // 4 accumulators: r/z merged (gi+gh), n split for the r*h_n term.
      // b_ih1 is folded into the x-WMMA via the constant-1 column.
      const float br = b_hh1[gcol], bz = b_hh1[gcol + HDIM],
                  bn = b_hh1[gcol + 2 * HDIM];
      v8f accr, accz, gin, ghn;
#pragma unroll
      for (int v = 0; v < 8; ++v) {
        accr[v] = br; accz[v] = bz; gin[v] = 0.0f; ghn[v] = bn;
      }

      // gi = [x|1|0..] @ packed(W_ih1|b_ih1): one K=32 WMMA per gate
      {
        v16bf ax = load_afrag(xbuf, m0 + mA, XSTR, 0, sel);
        v16bf bx_r = bp_ih1[((size_t)c) * 32 + lane];
        v16bf bx_z = bp_ih1[((size_t)(c + CCH)) * 32 + lane];
        v16bf bx_n = bp_ih1[((size_t)(c + 2 * CCH)) * 32 + lane];
        accr = wmma_bf16(ax, bx_r, accr);
        accz = wmma_bf16(ax, bx_z, accz);
        gin  = wmma_bf16(ax, bx_n, gin);
      }

      // gh = h1 @ W_hh1^T, K = 512
#pragma unroll 4
      for (int kt = 0; kt < KT; ++kt) {
        v16bf ah = load_afrag(h1c, m0 + mA, HDIM, kt * 32, sel);
        v16bf b_r = bp_hh1[((size_t)kt * NTT + c) * 32 + lane];
        v16bf b_z = bp_hh1[((size_t)kt * NTT + c + CCH) * 32 + lane];
        v16bf b_n = bp_hh1[((size_t)kt * NTT + c + 2 * CCH) * 32 + lane];
        accr = wmma_bf16(ah, b_r, accr);
        accz = wmma_bf16(ah, b_z, accz);
        ghn  = wmma_bf16(ah, b_n, ghn);
      }

#pragma unroll
      for (int v = 0; v < 8; ++v) {
        int m = m0 + v + 8 * sel;
        float r  = sigmoidf_(accr[v]);
        float z  = sigmoidf_(accz[v]);
        float nn = tanhf_(gin[v] + r * ghn[v]);
        float hp = (float)h1c[m * HDIM + gcol];
        h1n[m * HDIM + gcol] = (bf16_t)((1.0f - z) * nn + z * hp);
      }
    }
    __syncthreads();

    // ================= layer 2 =================
    for (int u = wave; u < NUNITS; u += NWAVE) {
      const int mt   = u & (MT - 1);
      const int c    = u / MT;
      const int m0   = mt * 16;
      const int gcol = c * 16 + n;

      const float br = b_ih2[gcol] + b_hh2[gcol];
      const float bz = b_ih2[gcol + HDIM] + b_hh2[gcol + HDIM];
      const float bi = b_ih2[gcol + 2 * HDIM];
      const float bh = b_hh2[gcol + 2 * HDIM];
      v8f accr, accz, in_, hn;
#pragma unroll
      for (int v = 0; v < 8; ++v) {
        accr[v] = br; accz[v] = bz; in_[v] = bi; hn[v] = bh;
      }

#pragma unroll 2
      for (int kt = 0; kt < KT; ++kt) {
        v16bf a1 = load_afrag(h1n, m0 + mA, HDIM, kt * 32, sel);
        v16bf a2 = load_afrag(h2c, m0 + mA, HDIM, kt * 32, sel);
        v16bf bi_r = bp_ih2[((size_t)kt * NTT + c) * 32 + lane];
        v16bf bi_z = bp_ih2[((size_t)kt * NTT + c + CCH) * 32 + lane];
        v16bf bi_n = bp_ih2[((size_t)kt * NTT + c + 2 * CCH) * 32 + lane];
        v16bf bh_r = bp_hh2[((size_t)kt * NTT + c) * 32 + lane];
        v16bf bh_z = bp_hh2[((size_t)kt * NTT + c + CCH) * 32 + lane];
        v16bf bh_n = bp_hh2[((size_t)kt * NTT + c + 2 * CCH) * 32 + lane];
        accr = wmma_bf16(a1, bi_r, accr);
        accr = wmma_bf16(a2, bh_r, accr);
        accz = wmma_bf16(a1, bi_z, accz);
        accz = wmma_bf16(a2, bh_z, accz);
        in_  = wmma_bf16(a1, bi_n, in_);
        hn   = wmma_bf16(a2, bh_n, hn);
      }

#pragma unroll
      for (int v = 0; v < 8; ++v) {
        int m = m0 + v + 8 * sel;
        float r  = sigmoidf_(accr[v]);
        float z  = sigmoidf_(accz[v]);
        float nn = tanhf_(in_[v] + r * hn[v]);
        float hp = (float)h2c[m * HDIM + gcol];
        h2n[m * HDIM + gcol] = (bf16_t)((1.0f - z) * nn + z * hp);
      }
    }
    __syncthreads();

    // ============ y = h2 @ W_out^T + b_out ; write out + next x ============
    if (wave < MT) {
      const int m0 = wave * 16;
      v8f acc;
      const float binit = (n < NOUT) ? b_out[n] : 0.0f;
#pragma unroll
      for (int v = 0; v < 8; ++v) acc[v] = binit;
#pragma unroll 4
      for (int kt = 0; kt < KT; ++kt) {
        v16bf af = load_afrag(h2n, m0 + mA, HDIM, kt * 32, sel);
        v16bf bw = bp_out[(size_t)kt * 32 + lane];
        acc = wmma_bf16(af, bw, acc);
      }
#pragma unroll
      for (int v = 0; v < 8; ++v) {
        int m = m0 + v + 8 * sel;
        if (n < NOUT) {
          out[((size_t)(bbase + m) * T_TOT + t) * NOUT + n] = acc[v];
          xbuf[m * XSTR + n] = (bf16_t)acc[v];
        }
      }
    } else if (wave == MT) {
      if (t + 1 < T_TOT) {
        for (int idx = lane; idx < BTILE * NCMD; idx += 32) {
          int r = idx / NCMD, i = idx - r * NCMD;
          xbuf[r * XSTR + NOUT + i] =
              (bf16_t)cmds[((size_t)(t + 1) * B_TOT + bbase + r) * NCMD + i];
        }
      }
    }
    __syncthreads();
  }
}

// ---------------------------------------------------------------------------
extern "C" void kernel_launch(void* const* d_in, const int* in_sizes, int n_in,
                              void* d_out, int out_size, void* d_ws,
                              size_t ws_size, hipStream_t stream) {
  (void)in_sizes; (void)n_in; (void)out_size; (void)ws_size;
  const float* cmds = (const float*)d_in[1];
  const float* y0   = (const float*)d_in[2];
  const float* h01  = (const float*)d_in[3];
  const float* h02  = (const float*)d_in[4];
  const float* Wih1 = (const float*)d_in[5];
  const float* Whh1 = (const float*)d_in[6];
  const float* bih1 = (const float*)d_in[7];
  const float* bhh1 = (const float*)d_in[8];
  const float* Wih2 = (const float*)d_in[9];
  const float* Whh2 = (const float*)d_in[10];
  const float* bih2 = (const float*)d_in[11];
  const float* bhh2 = (const float*)d_in[12];
  const float* Wout = (const float*)d_in[13];
  const float* bout = (const float*)d_in[14];
  float* out = (float*)d_out;

  bf16_t* pWhh1 = (bf16_t*)d_ws;
  bf16_t* pWih2 = pWhh1 + PW_ELEMS;
  bf16_t* pWhh2 = pWih2 + PW_ELEMS;
  bf16_t* pWout = pWhh2 + PW_ELEMS;
  bf16_t* pWih1 = pWout + PWOUT_ELEMS;

  // Pack weights into WMMA B-fragment layout (bf16).
  {
    int tot = KT * NTT * 32;
    int blk = 256, grd = (tot + blk - 1) / blk;
    pack_weight_kernel<<<grd, blk, 0, stream>>>(Whh1, pWhh1, HDIM);
    pack_weight_kernel<<<grd, blk, 0, stream>>>(Wih2, pWih2, HDIM);
    pack_weight_kernel<<<grd, blk, 0, stream>>>(Whh2, pWhh2, HDIM);
    pack_wout_kernel<<<(KT * 32 + 255) / 256, 256, 0, stream>>>(Wout, pWout);
    pack_wih1_kernel<<<(NTT * 32 + 255) / 256, 256, 0, stream>>>(Wih1, bih1,
                                                                 pWih1);
  }

  size_t lds_bytes = (size_t)4 * BTILE * HDIM * sizeof(bf16_t)  // h1+h2 x2 buf
                   + (size_t)BTILE * XSTR * sizeof(bf16_t);     // padded x
  (void)hipFuncSetAttribute((const void*)gru_persistent,
                            hipFuncAttributeMaxDynamicSharedMemorySize,
                            (int)lds_bytes);
  gru_persistent<<<B_TOT / BTILE, NTHREADS, lds_bytes, stream>>>(
      cmds, y0, h01, h02, bhh1, bih2, bhh2, bout,
      pWih1, pWhh1, pWih2, pWhh2, pWout, out);
}
